// MoeLayerTopKLayer_42932493091117
// MI455X (gfx1250) — matrix operand order repaired and checked by reading, not compiled, not run on hardware
//
#include <hip/hip_runtime.h>
#include <stdint.h>
#include <stddef.h>

// ---------------- problem constants ----------------
#define NUM_EXPERTS 8
#define TOPK        2
#define D_IN        1024
#define D_HID       4096
#define D_OUT       1024
#define T_TOKENS    8192
#define ASSIGN      (T_TOKENS * TOPK)          // 16384 (token,expert) assignments
#define MTILE       16                         // token rows per block tile (= TDM gather row limit)
#define MAX_TILES   (ASSIGN / MTILE + NUM_EXPERTS)  // 1032, worst-case tile count

// ---------------- CDNA5 WMMA / TDM types ----------------
typedef __attribute__((ext_vector_type(16))) __bf16       v16bf;
typedef __attribute__((ext_vector_type(8)))  float        v8f;
typedef __attribute__((ext_vector_type(4)))  unsigned int u32x4;
typedef __attribute__((ext_vector_type(4)))  unsigned int v4u;
typedef __attribute__((ext_vector_type(4)))  int          v4i;
typedef __attribute__((ext_vector_type(8)))  int          v8i;

union Frag { u32x4 q[2]; v16bf v; };   // 32 bytes: one 16x32-bf16 A/B fragment per lane

static __device__ inline v8f wmma_bf16(v16bf a, v16bf b, v8f c) {
  // D = A(16x32 bf16) * B(32x16 bf16) + C(16x16 f32)
  return __builtin_amdgcn_wmma_f32_16x16x32_bf16(false, a, false, b, (short)0, c,
                                                 false, false);
}

// A fragment (16x32, 16-bit): lane holds row m = lane&15; lane group kg = lane>>4.
// VGPR v<4  : K = kg*8 + 2v,2v+1      (K run [kg*8, kg*8+8))
// VGPR v>=4 : K = 16 + kg*8 + 2(v-4).. (K run [16+kg*8, 16+kg*8+8))
// -> two contiguous 8-element (16B) loads per lane.
static __device__ inline v16bf load_a(const __bf16* rowm, int k0, int kg) {
  Frag f;
  f.q[0] = *(const u32x4*)(rowm + k0 + kg * 8);
  f.q[1] = *(const u32x4*)(rowm + k0 + 16 + kg * 8);
  return f.v;
}

// B fragment (32x16, 16-bit): lane holds col n = lane&15; same K order as A:
// v<4: K = kg*8 + 2v,2v+1; v>=4: K = 16 + kg*8 + 2(v-4).. Weights are stored
// transposed (col-major), so each lane reads two 8-element (16B) K runs.
static __device__ inline v16bf load_b(const __bf16* __restrict__ Wt, int ldT,
                                      int k0, int n0, int lane) {
  int n = lane & 15, kg = lane >> 4;
  const __bf16* p = Wt + (size_t)(n0 + n) * ldT + k0 + kg * 8;
  Frag f;
  f.q[0] = *(const u32x4*)(p);
  f.q[1] = *(const u32x4*)(p + 16);
  return f.v;
}

// ---------------- workspace layout (bytes) ----------------
#define XB_OFF    ((size_t)0)                                   // x   bf16: 16 MiB
#define W1T_OFF   (XB_OFF  + (size_t)T_TOKENS * D_IN * 2)       // W1t bf16: 64 MiB
#define W2T_OFF   (W1T_OFF + (size_t)NUM_EXPERTS * D_IN * D_HID * 2)
#define SEL_OFF   (W2T_OFF + (size_t)NUM_EXPERTS * D_HID * D_OUT * 2)
#define SELW_OFF  (SEL_OFF  + (size_t)T_TOKENS * TOPK * 4)
#define CNT_OFF   (SELW_OFF + (size_t)T_TOKENS * TOPK * 4)
#define PSUM_OFF  (CNT_OFF  + 256)
#define LIST_OFF  (PSUM_OFF + 256)
#define PFX_OFF   (LIST_OFF + (size_t)NUM_EXPERTS * ASSIGN * 4)

// ---------------- kernels ----------------

__global__ void k_zero(float* out, long n_out, int* cnt, float* psum) {
  long i = (long)blockIdx.x * blockDim.x + threadIdx.x;
  long stride = (long)gridDim.x * blockDim.x;
  for (long j = i; j < n_out; j += stride) out[j] = 0.f;
  if (i < NUM_EXPERTS) { cnt[i] = 0; psum[i] = 0.f; }
}

// fp32 -> bf16 conversion; W1/W2 are transposed so GEMM B-fragments are
// contiguous: W1t[e][h][d] (ld=D_IN), W2t[e][o][h] (ld=D_HID).
__global__ void k_convert(const float* __restrict__ x, const float* __restrict__ W1,
                          const float* __restrict__ W2, __bf16* __restrict__ xb,
                          __bf16* __restrict__ W1t, __bf16* __restrict__ W2t) {
  const long NX = (long)T_TOKENS * D_IN;
  const long NW = (long)NUM_EXPERTS * D_IN * D_HID;  // same count for W2
  long i = (long)blockIdx.x * blockDim.x + threadIdx.x;
  long stride = (long)gridDim.x * blockDim.x;
  for (long j = i; j < NW; j += stride) {
    long e = j / ((long)D_IN * D_HID);
    long r = j % ((long)D_IN * D_HID);
    long d = r / D_HID, h = r % D_HID;       // W1 (e,d,h)
    W1t[e * (long)D_IN * D_HID + h * D_IN + d] = (__bf16)W1[j];
    long h2 = r / D_OUT, o = r % D_OUT;      // W2 (e,h,o)
    W2t[e * (long)D_HID * D_OUT + o * D_HID + h2] = (__bf16)W2[j];
    if (j < NX) xb[j] = (__bf16)x[j];
  }
}

// One wave (32 lanes) per token: logits, softmax, top-2, expert lists + stats.
__global__ void k_router(const float* __restrict__ x, const float* __restrict__ gw,
                         int* __restrict__ sel, float* __restrict__ selw,
                         int* __restrict__ cnt, float* __restrict__ psum,
                         int* __restrict__ lists) {
  int wave = threadIdx.x >> 5;
  int lane = threadIdx.x & 31;
  int t = blockIdx.x * (blockDim.x >> 5) + wave;
  if (t >= T_TOKENS) return;
  float acc[NUM_EXPERTS];
#pragma unroll
  for (int e = 0; e < NUM_EXPERTS; e++) acc[e] = 0.f;
  const float* xr = x + (size_t)t * D_IN;
  for (int d = lane; d < D_IN; d += 32) {
    float xv = xr[d];
    const float* g = gw + (size_t)d * NUM_EXPERTS;
#pragma unroll
    for (int e = 0; e < NUM_EXPERTS; e++) acc[e] += xv * g[e];
  }
#pragma unroll
  for (int e = 0; e < NUM_EXPERTS; e++)
    for (int off = 16; off; off >>= 1) acc[e] += __shfl_xor(acc[e], off, 32);
  if (lane == 0) {
    float mx = acc[0];
    for (int e = 1; e < NUM_EXPERTS; e++) mx = fmaxf(mx, acc[e]);
    float p[NUM_EXPERTS], s = 0.f;
    for (int e = 0; e < NUM_EXPERTS; e++) { p[e] = __expf(acc[e] - mx); s += p[e]; }
    for (int e = 0; e < NUM_EXPERTS; e++) p[e] /= s;
    int i0 = 0;
    for (int e = 1; e < NUM_EXPERTS; e++) if (p[e] > p[i0]) i0 = e;
    int i1 = (i0 == 0) ? 1 : 0;
    for (int e = 0; e < NUM_EXPERTS; e++) if (e != i0 && p[e] > p[i1]) i1 = e;
    float ws = p[i0] + p[i1] + 1e-6f;
    sel[t * 2] = i0;  sel[t * 2 + 1] = i1;
    selw[t * 2] = p[i0] / ws;  selw[t * 2 + 1] = p[i1] / ws;
    for (int e = 0; e < NUM_EXPERTS; e++) atomicAdd(&psum[e], p[e]);
    int s0 = atomicAdd(&cnt[i0], 1); lists[i0 * ASSIGN + s0] = t;
    int s1 = atomicAdd(&cnt[i1], 1); lists[i1 * ASSIGN + s1] = t;
  }
}

__global__ void k_prefix(const int* __restrict__ cnt, int* __restrict__ pfx) {
  if (threadIdx.x == 0 && blockIdx.x == 0) {
    int s = 0; pfx[0] = 0;
    for (int e = 0; e < NUM_EXPERTS; e++) {
      s += (cnt[e] + MTILE - 1) / MTILE;
      pfx[e + 1] = s;
    }
  }
}

// Fused expert MLP for one 16-token tile of one expert, 256 threads = 8 waves.
// Wave w owns output columns [w*128, w*128+128): 8 persistent 16x16 f32 tiles.
// X tile staging uses the Tensor Data Mover in gather mode: one DMA pulls the
// 16 gathered token rows (16-bit row indices, TDM gather limit = 16) into LDS.
__global__ void __launch_bounds__(256)
k_moe(const __bf16* __restrict__ xb, const __bf16* __restrict__ W1t,
      const __bf16* __restrict__ W2t, const float* __restrict__ b1,
      const float* __restrict__ b2, const int* __restrict__ lists,
      const int* __restrict__ cnt, const int* __restrict__ pfx,
      const int* __restrict__ sel, const float* __restrict__ selw,
      float* __restrict__ out) {
  __shared__ __bf16 Xs[MTILE][D_IN];   // 32 KB  : gathered token tile
  __shared__ __bf16 Hs[MTILE][128];    //  4 KB  : GELU'd hidden chunk
  __shared__ int   tokIdx[MTILE];
  __shared__ float tokW[MTILE];
  __shared__ int   sMeta[3];           // expert, row0, Ne

  int bx = blockIdx.x;
  if (threadIdx.x == 0) {
    int e = -1;
    for (int i = 0; i < NUM_EXPERTS; i++)
      if (bx >= pfx[i] && bx < pfx[i + 1]) { e = i; break; }
    sMeta[0] = e;
    sMeta[1] = (e >= 0) ? (bx - pfx[e]) * MTILE : 0;
    sMeta[2] = (e >= 0) ? cnt[e] : 0;
  }
  __syncthreads();
  int e = sMeta[0];
  if (e < 0) return;                   // uniform over the block
  int row0 = sMeta[1], Ne = sMeta[2];

  if (threadIdx.x < MTILE) {
    int r = threadIdx.x, ridx = row0 + r, tok = 0;
    float w = 0.f;
    if (ridx < Ne) {
      tok = lists[e * ASSIGN + ridx];
      w = (sel[tok * 2] == e) ? selw[tok * 2] : selw[tok * 2 + 1];
    }
    tokIdx[r] = tok; tokW[r] = w;
  }
  __syncthreads();

  int wave = threadIdx.x >> 5;
  int lane = threadIdx.x & 31;

  // ---- TDM gather: one tensor_load_to_lds for the 16-row X tile ----
  if (wave == 0) {
    uint64_t ga = (uint64_t)(uintptr_t)xb;            // tensor base (2D, 8192x1024 bf16)
    uint32_t lds = (uint32_t)(uintptr_t)&Xs[0][0];    // LDS byte address of tile
    v4u g0;
    g0[0] = 0x80000001u;                              // count=1, gather_mode=1, 16-bit idx
    g0[1] = lds;                                      // lds_addr [63:32]
    g0[2] = (uint32_t)ga;                             // global_addr [95:64]
    g0[3] = (uint32_t)((ga >> 32) & 0x01FFFFFFu) | (2u << 30);  // addr msbs + type=2
    v8i g1;
    g1[0] = (1 << 16);                 // wg_mask=0, data_size=1 (2 bytes)
    g1[1] = (int)((D_IN & 0xFFFF) << 16);      // tensor_dim0 = 1024 (bits 79:48)
    g1[2] = (int)((T_TOKENS & 0xFFFF) << 16);  // tensor_dim1 = 8192 (bits 111:80)
    g1[3] = (int)((D_IN & 0xFFFF) << 16);      // tile_dim0  = 1024 (bits 127:112)
    g1[4] = MTILE;                     // tile_dim1 = 16 valid gather indices
    g1[5] = D_IN;                      // tensor_dim0_stride = 1024 (bits 207:160)
    g1[6] = 0;
    g1[7] = 0;
    v4i g2, g3;                        // 16 packed 16-bit row indices
    g2[0] = tokIdx[0] | (tokIdx[1] << 16);
    g2[1] = tokIdx[2] | (tokIdx[3] << 16);
    g2[2] = tokIdx[4] | (tokIdx[5] << 16);
    g2[3] = tokIdx[6] | (tokIdx[7] << 16);
    g3[0] = tokIdx[8] | (tokIdx[9] << 16);
    g3[1] = tokIdx[10] | (tokIdx[11] << 16);
    g3[2] = tokIdx[12] | (tokIdx[13] << 16);
    g3[3] = tokIdx[14] | (tokIdx[15] << 16);
#if defined(__clang_major__) && __clang_major__ >= 23
    v8i gz = {0, 0, 0, 0, 0, 0, 0, 0};
    __builtin_amdgcn_tensor_load_to_lds(g0, g1, g2, g3, gz, 0);
#else
    __builtin_amdgcn_tensor_load_to_lds(g0, g1, g2, g3, 0);
#endif
    __builtin_amdgcn_s_wait_tensorcnt(0);
  }
  __syncthreads();

  int kg   = lane >> 4;
  int nloc = lane & 15;
  const __bf16* W1e = W1t + (size_t)e * D_HID * D_IN;   // [h][d], ld = D_IN
  const __bf16* W2e = W2t + (size_t)e * D_OUT * D_HID;  // [o][h], ld = D_HID
  const float*  b1e = b1 + (size_t)e * D_HID;
  const float*  b2e = b2 + (size_t)e * D_OUT;

  v8f accO[8];
#pragma unroll
  for (int t = 0; t < 8; t++) {
    float bv = b2e[wave * 128 + t * 16 + nloc];
    accO[t] = (v8f){bv, bv, bv, bv, bv, bv, bv, bv};
  }

  for (int hc = 0; hc < D_HID / 128; hc++) {
    // prefetch next chunk's weight tiles toward L2/WGP$ (global_prefetch_b8)
    if (hc + 1 < D_HID / 128) {
      __builtin_prefetch(W1e + (size_t)((hc + 1) * 128 + wave * 16 + nloc) * D_IN, 0, 1);
      __builtin_prefetch(W2e + (size_t)(wave * 128 + nloc * 8) * D_HID + (hc + 1) * 128, 0, 1);
    }
    // ---- GEMM1: 16x16 H tile per wave, K = D_IN ----
    float b1v = b1e[hc * 128 + wave * 16 + nloc];
    v8f accH = (v8f){b1v, b1v, b1v, b1v, b1v, b1v, b1v, b1v};
#pragma unroll 4
    for (int kk = 0; kk < D_IN / 32; kk++) {
      v16bf a = load_a(&Xs[nloc][0], kk * 32, kg);
      v16bf b = load_b(W1e, D_IN, kk * 32, hc * 128 + wave * 16, lane);
      accH = wmma_bf16(a, b, accH);
    }
    __syncthreads();  // prior chunk's Hs reads complete before overwrite
    // exact-erf GELU, C-tile layout: VGPR r -> M = r (lanes<16) / r+8, N = nloc
#pragma unroll
    for (int r = 0; r < 8; r++) {
      int m = (lane < 16) ? r : r + 8;
      float h = accH[r];
      h = 0.5f * h * (1.0f + erff(h * 0.70710678118654752f));
      Hs[m][wave * 16 + nloc] = (__bf16)h;
    }
    __syncthreads();
    // ---- GEMM2: accumulate wave's 16x128 out slice, K = 128 ----
#pragma unroll
    for (int kk = 0; kk < 4; kk++) {
      v16bf a = load_a(&Hs[nloc][0], kk * 32, kg);
#pragma unroll
      for (int t = 0; t < 8; t++) {
        v16bf b = load_b(W2e, D_HID, hc * 128 + kk * 32, wave * 128 + t * 16, lane);
        accO[t] = wmma_bf16(a, b, accO[t]);
      }
    }
  }

  // weighted scatter-add into out (each token receives exactly TOPK adds)
#pragma unroll
  for (int t = 0; t < 8; t++) {
    int col = wave * 128 + t * 16 + nloc;
#pragma unroll
    for (int r = 0; r < 8; r++) {
      int m = (lane < 16) ? r : r + 8;
      if (row0 + m < Ne)
        atomicAdd(&out[(size_t)tokIdx[m] * D_OUT + col], tokW[m] * accO[t][r]);
    }
  }
}

__global__ void k_aux(const int* __restrict__ cnt, const float* __restrict__ psum,
                      float* __restrict__ out_aux) {
  if (threadIdx.x == 0 && blockIdx.x == 0) {
    float s = 0.f;
    for (int e = 0; e < NUM_EXPERTS; e++)
      s += (psum[e] / (float)T_TOKENS) * ((float)cnt[e] / (float)(T_TOKENS * TOPK));
    out_aux[0] = (float)NUM_EXPERTS * s;
  }
}

// ---------------- host entry ----------------
extern "C" void kernel_launch(void* const* d_in, const int* in_sizes, int n_in,
                              void* d_out, int out_size, void* d_ws, size_t ws_size,
                              hipStream_t stream) {
  (void)in_sizes; (void)n_in; (void)ws_size;
  const float* x  = (const float*)d_in[0];
  const float* gw = (const float*)d_in[1];
  const float* W1 = (const float*)d_in[2];
  const float* b1 = (const float*)d_in[3];
  const float* W2 = (const float*)d_in[4];
  const float* b2 = (const float*)d_in[5];
  float* out = (float*)d_out;

  char* ws = (char*)d_ws;
  __bf16* xb   = (__bf16*)(ws + XB_OFF);
  __bf16* W1t  = (__bf16*)(ws + W1T_OFF);
  __bf16* W2t  = (__bf16*)(ws + W2T_OFF);
  int*    sel  = (int*)(ws + SEL_OFF);
  float*  selw = (float*)(ws + SELW_OFF);
  int*    cnt  = (int*)(ws + CNT_OFF);
  float*  psum = (float*)(ws + PSUM_OFF);
  int*    list = (int*)(ws + LIST_OFF);
  int*    pfx  = (int*)(ws + PFX_OFF);

  k_zero<<<2048, 256, 0, stream>>>(out, (long)out_size, cnt, psum);
  k_convert<<<4096, 256, 0, stream>>>(x, W1, W2, xb, W1t, W2t);
  k_router<<<T_TOKENS / 8, 256, 0, stream>>>(x, gw, sel, selw, cnt, psum, list);
  k_prefix<<<1, 32, 0, stream>>>(cnt, pfx);
  k_moe<<<MAX_TILES, 256, 0, stream>>>(xb, W1t, W2t, b1, b2, list, cnt, pfx,
                                       sel, selw, out);
  k_aux<<<1, 32, 0, stream>>>(cnt, psum, out + (size_t)T_TOKENS * D_OUT);
}
